// SAGE_77506979824092
// MI455X (gfx1250) — matrix operand order repaired and checked by reading, not compiled, not run on hardware
//
#include <hip/hip_runtime.h>

#define N_NODES 100000
#define N_EDGES 1600000
#define DIM 128
#define LDS_STRIDE 264   // 256 K-elements + padding (keeps 4B alignment, spreads banks)

typedef __attribute__((ext_vector_type(16))) _Float16 v16h;
typedef __attribute__((ext_vector_type(8)))  float    v8f;

__global__ void k_zero(float* p, int n) {
    int i = blockIdx.x * blockDim.x + threadIdx.x;
    int stride = gridDim.x * blockDim.x;
    for (; i < n; i += stride) p[i] = 0.0f;
}

__global__ void k_degree(const int* __restrict__ dst, float* __restrict__ deg) {
    int i = blockIdx.x * blockDim.x + threadIdx.x;
    if (i < N_EDGES) unsafeAtomicAdd(&deg[dst[i]], 1.0f);
}

__global__ void k_invert(float* deg) {
    int i = blockIdx.x * blockDim.x + threadIdx.x;
    if (i < N_NODES) deg[i] = 1.0f / fmaxf(deg[i], 1.0f);
}

// Pre-pack combined [wself; wneigh] (256x128 f32) into f16 WMMA B-fragment order:
// flat index o = ((kt*8 + wv)*32 + lane)*16 + elem, so each (wave, K-chunk) fragment
// is a contiguous, coalesced 1KB line; each lane's 16 f16 = one 32B vector load.
__global__ void k_pack_w(const float* __restrict__ wself,
                         const float* __restrict__ wneigh,
                         _Float16* __restrict__ out) {
    int o = blockIdx.x * blockDim.x + threadIdx.x;
    if (o >= 8 * 8 * 32 * 16) return;
    int elem = o & 15;
    int lane = (o >> 4) & 31;
    int wv   = (o >> 9) & 7;
    int kt   = o >> 12;
    int l16 = lane & 15;
    int lh  = lane >> 4;
    int n = wv * 16 + l16;                       // output column
    int k = kt * 32 + ((elem >= 8) ? 16 : 0) + lh * 8 + (elem & 7);  // = kt*32 + (elem>=8?16:0) + lh*8 + (elem&7)
    float v = (k < 128) ? wself[k * DIM + n] : wneigh[(k - 128) * DIM + n];
    out[o] = (_Float16)v;
}

// One wave per edge: 32 lanes x float4 = 128 floats. Gather h[src], fp32-atomic-add into agg[dst].
// agg (51.2 MB) is L2-resident (192 MB L2), so these atomics land in L2.
__global__ __launch_bounds__(256) void k_scatter(const float* __restrict__ h,
                                                 const int* __restrict__ src,
                                                 const int* __restrict__ dst,
                                                 float* __restrict__ agg) {
    int edge = (int)((blockIdx.x * blockDim.x + threadIdx.x) >> 5);
    int lane = threadIdx.x & 31;
    if (edge >= N_EDGES) return;
    int s = src[edge];
    int d = dst[edge];
    const float4* hv = (const float4*)(h + (size_t)s * DIM);
    float4 v = hv[lane];
    float* a = agg + (size_t)d * DIM + lane * 4;
    unsafeAtomicAdd(a + 0, v.x);
    unsafeAtomicAdd(a + 1, v.y);
    unsafeAtomicAdd(a + 2, v.z);
    unsafeAtomicAdd(a + 3, v.w);
}

// Fused SAGE layer: out = act( [feat | agg*inv_deg] @ [wself ; wneigh] + bias )
// Block = 256 threads (8 waves). Block computes 32 rows x 128 cols.
// Wave w owns N-tile [16w, 16w+16); computes two 16x16 D tiles via v_wmma_f32_16x16x32_f16.
// A (K=256, f16) staged in LDS; B fragments are pre-packed f16 -> one 32B load per lane per kt.
template <bool RELU>
__global__ __launch_bounds__(256) void k_sage_gemm(const float* feat,
                                                   const float* __restrict__ agg,
                                                   const float* __restrict__ inv_deg,
                                                   const v16h* __restrict__ wpk,
                                                   const float* __restrict__ bias,
                                                   float* out) {
    __shared__ _Float16 At[32 * LDS_STRIDE];

    const int tid = threadIdx.x;
    const int m0  = blockIdx.x * 32;

    // Stage A tile: rows m0..m0+31, K-cols [0,128)=feat, [128,256)=agg*inv_deg, as f16.
    for (int idx = tid; idx < 32 * 256; idx += 256) {
        int r = idx >> 8;
        int k = idx & 255;
        int m = m0 + r;
        float v;
        if (k < 128) v = feat[(size_t)m * DIM + k];
        else         v = agg[(size_t)m * DIM + (k - 128)] * inv_deg[m];
        At[r * LDS_STRIDE + k] = (_Float16)v;
    }
    __syncthreads();

    const int wv   = tid >> 5;   // wave id = N-tile index (0..7)
    const int lane = tid & 31;
    const int l16  = lane & 15;
    const int lh   = lane >> 4;  // lane half
    const int col  = wv * 16 + l16;

    v8f c0 = {};
    v8f c1 = {};

    #pragma unroll
    for (int kt = 0; kt < 8; ++kt) {        // K chunks of 32
        // ---- B fragment (32x16 f16): pre-packed, contiguous per lane.
        v16h b = wpk[(kt * 8 + wv) * 32 + lane];
        // ---- A fragments (16x32 f16), ISA 16-bit A layout:
        // lane half lh selects K sub-range: VGPR j -> K = (j>=4?16:0) + lh*8 + (j&3)*2 (+k0)
        const int k0 = kt * 32;
        v16h a0, a1;
        #pragma unroll
        for (int j = 0; j < 8; ++j) {
            int k = k0 + ((j >= 4) ? 16 : 0) + lh * 8 + (j & 3) * 2;
            a0[2 * j]     = At[l16        * LDS_STRIDE + k];
            a0[2 * j + 1] = At[l16        * LDS_STRIDE + k + 1];
            a1[2 * j]     = At[(16 + l16) * LDS_STRIDE + k];
            a1[2 * j + 1] = At[(16 + l16) * LDS_STRIDE + k + 1];
        }
        c0 = __builtin_amdgcn_wmma_f32_16x16x32_f16(false, a0, false, b, (short)0, c0, false, false);
        c1 = __builtin_amdgcn_wmma_f32_16x16x32_f16(false, a1, false, b, (short)0, c1, false, false);
    }

    // Epilogue: D layout -> row = lh*8 + i, col = l16 within each 16x16 tile.
    const float bv = bias[col];
    #pragma unroll
    for (int i = 0; i < 8; ++i) {
        int   r0 = m0 + lh * 8 + i;
        int   r1 = m0 + 16 + lh * 8 + i;
        float v0 = c0[i] + bv;
        float v1 = c1[i] + bv;
        if (RELU) { v0 = fmaxf(v0, 0.0f); v1 = fmaxf(v1, 0.0f); }
        out[(size_t)r0 * DIM + col] = v0;
        out[(size_t)r1 * DIM + col] = v1;
    }
}

extern "C" void kernel_launch(void* const* d_in, const int* in_sizes, int n_in,
                              void* d_out, int out_size, void* d_ws, size_t ws_size,
                              hipStream_t stream) {
    const float* x    = (const float*)d_in[0];
    const int*   esrc = (const int*)d_in[1];
    const int*   edst = (const int*)d_in[2];
    const float* ws1  = (const float*)d_in[3];
    const float* wn1  = (const float*)d_in[4];
    const float* b1   = (const float*)d_in[5];
    const float* ws2  = (const float*)d_in[6];
    const float* wn2  = (const float*)d_in[7];
    const float* b2   = (const float*)d_in[8];
    float* out = (float*)d_out;

    // Workspace layout (all offsets 32B-aligned):
    //   deg  : N_NODES f32 (becomes inv_deg)
    //   agg  : N_NODES*DIM f32 (51.2 MB, L2-resident)
    //   wpk1 : 32768 f16 packed B-fragments layer 1
    //   wpk2 : 32768 f16 packed B-fragments layer 2
    float* deg = (float*)d_ws;
    float* agg = deg + ((N_NODES + 255) & ~255);
    _Float16* wpk1 = (_Float16*)(agg + (size_t)N_NODES * DIM);
    _Float16* wpk2 = wpk1 + 32768;
    const int NPK = 8 * 8 * 32 * 16;   // 32768 packed elements per layer

    // Degree + inv_deg + packed weights (computed fresh every call; d_ws is poisoned).
    k_zero<<<2048, 256, 0, stream>>>(deg, N_NODES);
    k_zero<<<8192, 256, 0, stream>>>(agg, N_NODES * DIM);
    k_degree<<<(N_EDGES + 255) / 256, 256, 0, stream>>>(edst, deg);
    k_invert<<<(N_NODES + 255) / 256, 256, 0, stream>>>(deg);
    k_pack_w<<<NPK / 256, 256, 0, stream>>>(ws1, wn1, wpk1);
    k_pack_w<<<NPK / 256, 256, 0, stream>>>(ws2, wn2, wpk2);

    // Layer 1: aggregate x, then fused dual-GEMM + bias + ReLU -> hidden state in d_out.
    k_scatter<<<N_EDGES / 8, 256, 0, stream>>>(x, esrc, edst, agg);
    k_sage_gemm<true><<<N_NODES / 32, 256, 0, stream>>>(x, agg, deg, (const v16h*)wpk1, b1, out);

    // Layer 2: re-aggregate hidden state, fused dual-GEMM + bias -> final output in d_out.
    // (Each GEMM block reads its 32 rows into LDS before overwriting them; rows are
    //  block-disjoint, so in-place h->out is safe.)
    k_zero<<<8192, 256, 0, stream>>>(agg, N_NODES * DIM);
    k_scatter<<<N_EDGES / 8, 256, 0, stream>>>(out, esrc, edst, agg);
    k_sage_gemm<false><<<N_NODES / 32, 256, 0, stream>>>(out, agg, deg, (const v16h*)wpk2, b2, out);
}
